// Stage_18717467476477
// MI455X (gfx1250) — matrix operand order repaired and checked by reading, not compiled, not run on hardware
//
#include <hip/hip_runtime.h>
#include <hip/hip_bf16.h>

// ---------------------------------------------------------------------------
// GCN/graph-style pipeline for MI455X (gfx1250, wave32, WMMA).
// All GEMMs: bf16 operands, f32 WMMA accumulation (v_wmma_f32_16x16x32_bf16).
// att stored bf16 (dominant traffic ~79MB, read 5x). fp32 for BN stats,
// degree rsqrt, residual stream.
// GEMM convention: A row-major [M][K], B passed transposed as Bt [N][K].
// All N-dimension row counts padded 3136 -> 3200 so the LDS staging is
// branch-free. Double-buffered LDS with CDNA5 async global->LDS copies
// (global_load_async_to_lds_b128, ASYNCcnt) -- no VGPR round trip.
// Transposed bf16 epilogue packs each 8-element column run into one B128.
// ---------------------------------------------------------------------------

typedef __attribute__((ext_vector_type(16))) __bf16 v16bf;
typedef __attribute__((ext_vector_type(8)))  float  v8f;

#define Bn 4
#define Cc 256
#define Pp 128
#define Nn 3136          // 56*56
#define NP 3200          // Nn padded to multiple of 128 (tile rows)
#define Ss 5
#define EPSBN 1e-5f

#define TM 128
#define TN 128
#define TK 32
#define KP 40            // padded LDS K-stride (80B rows: 16B aligned, bank-spread)

typedef unsigned long long ull;

union FragU { uint4 q[2]; v16bf v; };

// CDNA5 async global->LDS B128 copy (per-lane), tracked with ASYNCcnt.
__device__ __forceinline__ void async_ld_b128(unsigned lds_addr, const void* gptr) {
    asm volatile("global_load_async_to_lds_b128 %0, %1, off"
                 :: "v"(lds_addr), "v"(gptr) : "memory");
}
__device__ __forceinline__ void wait_async0() {
    asm volatile("s_wait_asynccnt 0x0" ::: "memory");
}
__device__ __forceinline__ unsigned lds_addr_of(const void* p) {
    return (unsigned)(size_t)p;   // addrspace(3) ptrtoint -> LDS byte address
}

// ---------------------------------------------------------------------------
// Universal bf16 WMMA GEMM: D = (A x Bt^T) [+ A2 x Bt2^T]
//   A  : [M][K] bf16, batch stride sA (0 for shared weights)
//   Bt : [N][K] bf16, batch stride sB
// Row counts of A/Bt buffers must be padded to multiples of 128 (no load
// guards). Outputs (any subset, epilogue-guarded):
//   outF  f32  row-major [M][N], outBr bf16 row-major, outBt bf16 [N][M].
// Block = 256 thr (8 waves, 2x4), block tile 128x128, wave tile 64x32.
// Double-buffered LDS fed by async global->LDS B128; 1 barrier / K-step.
// ---------------------------------------------------------------------------
__global__ __launch_bounds__(256)
void gemm_bf16_wmma(const __bf16* __restrict__ A,  const __bf16* __restrict__ A2,
                    const __bf16* __restrict__ Bt, const __bf16* __restrict__ Bt2,
                    ull sA, ull sB,
                    float* outF, ull sF,
                    __bf16* outBr, ull sBr,
                    __bf16* outBt, ull sBt,
                    int M, int N, int K, int relu)
{
    __shared__ __bf16 lA[2][TM * KP];
    __shared__ __bf16 lB[2][TN * KP];

    const int tid  = threadIdx.x;
    const int lane = tid & 31;
    const int wave = tid >> 5;
    const int wm   = wave >> 2;          // 0..1  (64-row band)
    const int wn   = wave & 3;           // 0..3  (32-col band)
    const int r    = lane & 15;
    const int h    = lane >> 4;

    const int m0 = blockIdx.y * TM;
    const int n0 = blockIdx.x * TN;
    const int b  = blockIdx.z;

    const __bf16* Ab   = A  + (size_t)b * sA;
    const __bf16* A2b  = A2 ? A2 + (size_t)b * sA : (const __bf16*)0;
    const __bf16* Btb  = Bt + (size_t)b * sB;
    const __bf16* Bt2b = Bt2 ? Bt2 + (size_t)b * sB : (const __bf16*)0;

    // per-thread staging coordinates: chunks tid and tid+256 of 512 uint4
    const int aRow = tid >> 2;           // 0..63 (second chunk: +64)
    const int aKq  = (tid & 3) * 8;      // 0,8,16,24
    const size_t gOffA0 = (size_t)(m0 + aRow)      * K + aKq;
    const size_t gOffA1 = (size_t)(m0 + aRow + 64) * K + aKq;
    const size_t gOffB0 = (size_t)(n0 + aRow)      * K + aKq;
    const size_t gOffB1 = (size_t)(n0 + aRow + 64) * K + aKq;
    const int lOff0 = aRow * KP + aKq;
    const int lOff1 = (aRow + 64) * KP + aKq;

    v8f acc[4][2];
    for (int mt = 0; mt < 4; ++mt)
        for (int nt = 0; nt < 2; ++nt)
            for (int e = 0; e < 8; ++e) acc[mt][nt][e] = 0.f;

    const int npass = A2b ? 2 : 1;
    const int total = npass * (K / TK);

    const __bf16* curA = Ab;
    const __bf16* curB = Btb;

    // prologue: async-copy tile 0 -> LDS[0]
    async_ld_b128(lds_addr_of(&lA[0][lOff0]), curA + gOffA0);
    async_ld_b128(lds_addr_of(&lA[0][lOff1]), curA + gOffA1);
    async_ld_b128(lds_addr_of(&lB[0][lOff0]), curB + gOffB0);
    async_ld_b128(lds_addr_of(&lB[0][lOff1]), curB + gOffB1);

    int k0 = 0, buf = 0, passIdx = 0;
    for (int it = 0; it < total; ++it) {
        wait_async0();       // my asyncs into LDS[buf] complete
        __syncthreads();     // all waves' asyncs complete; LDS[buf^1] free

        // next-tile coordinates
        int nk0 = k0 + TK;
        const __bf16* nA = curA;
        const __bf16* nB = curB;
        bool have = true;
        if (nk0 >= K) {
            nk0 = 0; ++passIdx;
            nA = A2b; nB = Bt2b;
            if (passIdx >= npass) have = false;
        }
        if (have) {
            int nb = buf ^ 1;
            async_ld_b128(lds_addr_of(&lA[nb][lOff0]), nA + gOffA0 + nk0);
            async_ld_b128(lds_addr_of(&lA[nb][lOff1]), nA + gOffA1 + nk0);
            async_ld_b128(lds_addr_of(&lB[nb][lOff0]), nB + gOffB0 + nk0);
            async_ld_b128(lds_addr_of(&lB[nb][lOff1]), nB + gOffB1 + nk0);
        }

        // ---- fragments from LDS[buf] ----
        // B frag: lane (h,r) -> col = r, K = h*16 .. h*16+15 (contiguous)
        FragU bfrag[2];
#pragma unroll
        for (int nt = 0; nt < 2; ++nt) {
            const __bf16* src = &lB[buf][(wn * 32 + nt * 16 + r) * KP + h * 16];
            bfrag[nt].q[0] = *(const uint4*)(src);
            bfrag[nt].q[1] = *(const uint4*)(src + 8);
        }
        // A frag: lane (h,r) -> row = r, K = h*16 .. h*16+15 (contiguous)
        FragU afrag[4];
#pragma unroll
        for (int mt = 0; mt < 4; ++mt) {
            const __bf16* src = &lA[buf][(wm * 64 + mt * 16 + r) * KP + h * 16];
            afrag[mt].q[0] = *(const uint4*)(src);
            afrag[mt].q[1] = *(const uint4*)(src + 8);
        }

#pragma unroll
        for (int mt = 0; mt < 4; ++mt)
#pragma unroll
            for (int nt = 0; nt < 2; ++nt)
                acc[mt][nt] = __builtin_amdgcn_wmma_f32_16x16x32_bf16(
                    false, afrag[mt].v, false, bfrag[nt].v,
                    (short)0, acc[mt][nt], false, false);

        curA = nA; curB = nB; k0 = nk0; buf ^= 1;
    }

    // ---- epilogue: C/D layout -> row = mt*16 + e + 8*h, col = nt*16 + r ----
    const bool vecT = (outBt != nullptr) && ((M & 15) == 0);
    if (vecT) {
        // transposed bf16 store: 8 consecutive rows per lane -> one B128
#pragma unroll
        for (int mt = 0; mt < 4; ++mt)
#pragma unroll
            for (int nt = 0; nt < 2; ++nt) {
                int col = n0 + wn * 32 + nt * 16 + r;
                if (col < N) {
                    union { __bf16 e8[8]; uint4 q; } pk;
#pragma unroll
                    for (int e = 0; e < 8; ++e) {
                        float v = acc[mt][nt][e];
                        if (relu) v = fmaxf(v, 0.f);
                        pk.e8[e] = (__bf16)v;
                    }
                    int row0 = m0 + wm * 64 + mt * 16 + 8 * h;
                    *(uint4*)&outBt[(size_t)b * sBt + (size_t)col * M + row0] = pk.q;
                }
            }
    }
    if (outF || outBr || (outBt && !vecT)) {
#pragma unroll
        for (int mt = 0; mt < 4; ++mt)
#pragma unroll
            for (int nt = 0; nt < 2; ++nt)
#pragma unroll
                for (int e = 0; e < 8; ++e) {
                    int row = m0 + wm * 64 + mt * 16 + e + 8 * h;
                    int col = n0 + wn * 32 + nt * 16 + r;
                    if (row < M && col < N) {
                        float v = acc[mt][nt][e];
                        if (relu) v = fmaxf(v, 0.f);
                        if (outF)  outF [(size_t)b * sF  + (size_t)row * N + col] = v;
                        if (outBr) outBr[(size_t)b * sBr + (size_t)row * N + col] = (__bf16)v;
                        if (outBt && !vecT)
                            outBt[(size_t)b * sBt + (size_t)col * M + row] = (__bf16)v;
                    }
                }
    }
}

// ---------------------------------------------------------------------------
// Elementwise / reduction helpers
// ---------------------------------------------------------------------------
__global__ void k_f32_to_bf16(const float* __restrict__ s, __bf16* __restrict__ d, int n) {
    int i = blockIdx.x * 256 + threadIdx.x;
    if (i < n) d[i] = (__bf16)s[i];
}

// x [B][C][N] f32 -> out_f32 copy (residual stream) + outT [B][NP][C] bf16
__global__ void k_prep_x(const float* __restrict__ x, float* __restrict__ outF,
                         __bf16* __restrict__ outT) {
    int i = blockIdx.x * 256 + threadIdx.x;
    if (i >= Bn * Cc * Nn) return;
    int n = i % Nn; int t = i / Nn; int c = t % Cc; int b = t / Cc;
    float v = x[i];
    outF[i] = v;
    outT[((size_t)b * NP + n) * Cc + c] = (__bf16)v;
}

// att = 0.5*(att + att^T), in place, bf16 ([B][NP rows][Nn cols])
__global__ void k_sym(__bf16* att) {
    int i = blockIdx.x * 256 + threadIdx.x;
    if (i >= Bn * Nn * Nn) return;
    int j = i % Nn; int t = i / Nn; int ii = t % Nn; int b = t / Nn;
    if (j <= ii) return;
    size_t base = (size_t)b * NP * Nn;
    float a0 = (float)att[base + (size_t)ii * Nn + j];
    float a1 = (float)att[base + (size_t)j  * Nn + ii];
    __bf16 v = (__bf16)(0.5f * (a0 + a1));
    att[base + (size_t)ii * Nn + j] = v;
    att[base + (size_t)j  * Nn + ii] = v;
}

// d_inv[b,i] = d!=0 ? rsqrt(sum_j att[b,i,j]) : 0   (b128-chunked row sum)
__global__ void k_degree(const __bf16* __restrict__ att, float* __restrict__ dinv) {
    int bi = blockIdx.x;                  // 0 .. B*Nn-1
    int b = bi / Nn, ii = bi % Nn;
    const __bf16* rowp = att + ((size_t)b * NP + ii) * Nn;
    float s = 0.f;
    for (int j0 = threadIdx.x * 8; j0 < Nn; j0 += 256 * 8) {
        uint4 v = *(const uint4*)(rowp + j0);
        const __bf16* e = (const __bf16*)&v;
#pragma unroll
        for (int u = 0; u < 8; ++u) s += (float)e[u];
    }
    __shared__ float red[256];
    red[threadIdx.x] = s; __syncthreads();
    for (int o = 128; o > 0; o >>= 1) {
        if (threadIdx.x < o) red[threadIdx.x] += red[threadIdx.x + o];
        __syncthreads();
    }
    if (threadIdx.x == 0) {
        float d = red[0];
        dinv[bi] = (d != 0.f) ? rsqrtf(d) : 0.f;
    }
}

// att[b,i,j] *= dinv[b,i]*dinv[b,j], in place, 8 elems (b128) per thread
__global__ void k_norm(__bf16* att, const float* __restrict__ dinv) {
    const int chunks = Nn / 8;            // 392
    int gid = blockIdx.x * 256 + threadIdx.x;
    if (gid >= Bn * Nn * chunks) return;
    int jc = gid % chunks; int t = gid / chunks; int ii = t % Nn; int b = t / Nn;
    size_t off = ((size_t)b * NP + ii) * Nn + jc * 8;
    uint4 v = *(uint4*)(att + off);
    __bf16* e = (__bf16*)&v;
    float di = dinv[b * Nn + ii];
    const float* dj = dinv + b * Nn + jc * 8;
#pragma unroll
    for (int u = 0; u < 8; ++u) e[u] = (__bf16)((float)e[u] * di * dj[u]);
    *(uint4*)(att + off) = v;
}

// per-channel mean/var over (b, n) of o [B][C][N]
__global__ void k_bnstat(const float* __restrict__ o, float* __restrict__ stats) {
    int c = blockIdx.x;
    float s1 = 0.f, s2 = 0.f;
    for (int t = threadIdx.x; t < Bn * Nn; t += 256) {
        int b = t / Nn, n = t % Nn;
        float v = o[((size_t)b * Cc + c) * Nn + n];
        s1 += v; s2 += v * v;
    }
    __shared__ float r1[256], r2[256];
    r1[threadIdx.x] = s1; r2[threadIdx.x] = s2; __syncthreads();
    for (int off = 128; off > 0; off >>= 1) {
        if (threadIdx.x < off) { r1[threadIdx.x] += r1[threadIdx.x + off];
                                 r2[threadIdx.x] += r2[threadIdx.x + off]; }
        __syncthreads();
    }
    if (threadIdx.x == 0) {
        float inv = 1.f / (float)(Bn * Nn);
        float m = r1[0] * inv;
        stats[c]      = m;
        stats[Cc + c] = r2[0] * inv - m * m;
    }
}

// out = BN(o)*gamma + beta + out_prev ; also refresh transposed bf16 stream
__global__ void k_bnapply(const float* __restrict__ o, const float* __restrict__ outPrev,
                          const float* __restrict__ stats,
                          const float* __restrict__ gamma, const float* __restrict__ beta,
                          float* __restrict__ outNext, __bf16* __restrict__ outT) {
    int i = blockIdx.x * 256 + threadIdx.x;
    if (i >= Bn * Cc * Nn) return;
    int n = i % Nn; int t = i / Nn; int c = t % Cc; int b = t / Cc;
    float m = stats[c], var = stats[Cc + c];
    float v = (o[i] - m) * rsqrtf(var + EPSBN) * gamma[c] + beta[c] + outPrev[i];
    outNext[i] = v;
    outT[((size_t)b * NP + n) * Cc + c] = (__bf16)v;
}

// ---------------------------------------------------------------------------
extern "C" void kernel_launch(void* const* d_in, const int* in_sizes, int n_in,
                              void* d_out, int out_size, void* d_ws, size_t ws_size,
                              hipStream_t stream) {
    (void)in_sizes; (void)n_in; (void)out_size; (void)ws_size;
    const float* x     = (const float*)d_in[0];
    const float* Wt    = (const float*)d_in[1];
    const float* Wp    = (const float*)d_in[2];
    const float* Wg    = (const float*)d_in[3];
    const float* W1    = (const float*)d_in[4];
    const float* W2    = (const float*)d_in[5];
    const float* gamma = (const float*)d_in[6];
    const float* beta  = (const float*)d_in[7];
    float* out = (float*)d_out;

    char* w = (char*)d_ws;
    auto alloc = [&](size_t bytes) -> char* {
        char* p = w; w += (bytes + 255) & ~(size_t)255; return p;
    };
    __bf16* att  = (__bf16*)alloc((size_t)Bn * NP * Nn * 2);   // 80.3 MB
    __bf16* outT = (__bf16*)alloc((size_t)Bn * NP * Cc * 2);
    float*  outF = (float*) alloc((size_t)Bn * Cc * Nn * 4);
    __bf16* tT   = (__bf16*)alloc((size_t)Bn * NP * Pp * 2);
    __bf16* pT   = (__bf16*)alloc((size_t)Bn * NP * Pp * 2);
    __bf16* gR   = (__bf16*)alloc((size_t)Bn * Pp * Nn * 2);
    __bf16* gT   = (__bf16*)alloc((size_t)Bn * NP * Pp * 2);
    __bf16* agT  = (__bf16*)alloc((size_t)Bn * NP * Pp * 2);
    float*  oF   = (float*) alloc((size_t)Bn * Cc * Nn * 4);
    float*  dinv = (float*) alloc((size_t)Bn * Nn * 4);
    float*  stats= (float*) alloc((size_t)2 * Cc * 4);
    __bf16* wtb  = (__bf16*)alloc((size_t)Pp * Cc * 2);
    __bf16* wpb  = (__bf16*)alloc((size_t)Pp * Cc * 2);
    __bf16* wgb  = (__bf16*)alloc((size_t)Ss * Pp * Cc * 2);
    __bf16* w1b  = (__bf16*)alloc((size_t)Ss * Cc * Pp * 2);
    __bf16* w2b  = (__bf16*)alloc((size_t)Ss * Cc * Pp * 2);

    const int EWB = 256;
    const int nPC = Pp * Cc, nSPC = Ss * Pp * Cc, nBCN = Bn * Cc * Nn, nAtt = Bn * Nn * Nn;

    k_f32_to_bf16<<<(nPC  + EWB - 1) / EWB, EWB, 0, stream>>>(Wt, wtb, nPC);
    k_f32_to_bf16<<<(nPC  + EWB - 1) / EWB, EWB, 0, stream>>>(Wp, wpb, nPC);
    k_f32_to_bf16<<<(nSPC + EWB - 1) / EWB, EWB, 0, stream>>>(Wg, wgb, nSPC);
    k_f32_to_bf16<<<(nSPC + EWB - 1) / EWB, EWB, 0, stream>>>(W1, w1b, nSPC);
    k_f32_to_bf16<<<(nSPC + EWB - 1) / EWB, EWB, 0, stream>>>(W2, w2b, nSPC);
    k_prep_x<<<(nBCN + EWB - 1) / EWB, EWB, 0, stream>>>(x, outF, outT);

    dim3 blk(256);
    dim3 gProj(NP / TN, (Pp + TM - 1) / TM, Bn);   // 25 x 1 x 4
    dim3 gAtt (NP / TN, NP / TM, Bn);              // 25 x 25 x 4
    dim3 gO   (NP / TN, (Cc + TM - 1) / TM, Bn);   // 25 x 2 x 4

    // t = relu(Wt @ xf)  -> tT [B][NP][P];  p likewise
    gemm_bf16_wmma<<<gProj, blk, 0, stream>>>(wtb, nullptr, outT, nullptr,
        0ULL, (ull)NP * Cc, nullptr, 0ULL, nullptr, 0ULL, tT, (ull)NP * Pp,
        Pp, Nn, Cc, 1);
    gemm_bf16_wmma<<<gProj, blk, 0, stream>>>(wpb, nullptr, outT, nullptr,
        0ULL, (ull)NP * Cc, nullptr, 0ULL, nullptr, 0ULL, pT, (ull)NP * Pp,
        Pp, Nn, Cc, 1);
    // att = t^T p  (A = tT [NP][P], Bt = pT [NP][P]) -> bf16 row-major
    gemm_bf16_wmma<<<gAtt, blk, 0, stream>>>(tT, nullptr, pT, nullptr,
        (ull)NP * Pp, (ull)NP * Pp, nullptr, 0ULL, att, (ull)NP * Nn,
        nullptr, 0ULL, Nn, Nn, Pp, 0);

    k_sym   <<<(nAtt + EWB - 1) / EWB, EWB, 0, stream>>>(att);
    k_degree<<<Bn * Nn, EWB, 0, stream>>>(att, dinv);
    {
        int nch = Bn * Nn * (Nn / 8);
        k_norm<<<(nch + EWB - 1) / EWB, EWB, 0, stream>>>(att, dinv);
    }

    for (int s = 0; s < Ss; ++s) {
        // g = Wg[s] @ out   -> gR [B][P][N] and gT [B][NP][P]
        gemm_bf16_wmma<<<gProj, blk, 0, stream>>>(wgb + (size_t)s * Pp * Cc, nullptr,
            outT, nullptr, 0ULL, (ull)NP * Cc,
            nullptr, 0ULL, gR, (ull)Pp * Nn, gT, (ull)NP * Pp,
            Pp, Nn, Cc, 0);
        // ag = g @ att (att symmetric) -> agT [B][NP][P]
        gemm_bf16_wmma<<<gProj, blk, 0, stream>>>(gR, nullptr, att, nullptr,
            (ull)Pp * Nn, (ull)NP * Nn,
            nullptr, 0ULL, nullptr, 0ULL, agT, (ull)NP * Pp,
            Pp, Nn, Nn, 0);
        // o = W1[s] @ ag + W2[s] @ g  (dual-pass fused) -> f32 [B][C][N]
        gemm_bf16_wmma<<<gO, blk, 0, stream>>>(w1b + (size_t)s * Cc * Pp,
            w2b + (size_t)s * Cc * Pp, agT, gT,
            0ULL, (ull)NP * Pp,
            oF, (ull)Cc * Nn, nullptr, 0ULL, nullptr, 0ULL,
            Cc, Nn, Pp, 0);
        // BN stats + apply + residual (+ refresh transposed bf16 stream)
        k_bnstat<<<Cc, EWB, 0, stream>>>(oF, stats);
        float* dst = (s == Ss - 1) ? out : outF;
        k_bnapply<<<(nBCN + EWB - 1) / EWB, EWB, 0, stream>>>(
            oF, outF, stats, gamma + s * Cc, beta + s * Cc, dst, outT);
    }
}
